// MultiHeadSelfAttention_64355789963338
// MI455X (gfx1250) — matrix operand order repaired and checked by reading, not compiled, not run on hardware
//
#include <hip/hip_runtime.h>

// ---------------------------------------------------------------------------
// MHA forward for MI455X (gfx1250, wave32, WMMA).
// Compute-bound (~45 GFLOP vs ~2us of HBM traffic) -> bf16 WMMA everywhere,
// f32 accumulate. Flash-attention structure: no LxL score materialization.
// Streaming operands prefetched one tile ahead (global_prefetch_b8,
// unconditional: speculative prefetch past the end is silently dropped).
// ---------------------------------------------------------------------------

typedef __bf16 bf16;
typedef __attribute__((ext_vector_type(8)))  bf16  bf16x8;
typedef __attribute__((ext_vector_type(16))) bf16  bf16x16;
typedef __attribute__((ext_vector_type(8)))  float f32x8;

#define BB   2
#define LL   2048
#define DD   768
#define HH   12
#define HD   64
#define SCL  0.125f   // 64^-0.5

static __device__ __forceinline__ bf16x16 cat8(bf16x8 lo, bf16x8 hi) {
    return __builtin_shufflevector(lo, hi, 0,1,2,3,4,5,6,7,8,9,10,11,12,13,14,15);
}

static __device__ __forceinline__ f32x8 wmma_bf16(bf16x16 a, bf16x16 b, f32x8 c) {
    // D = A(16x32 bf16) x B(32x16 bf16) + C(16x16 f32)
    return __builtin_amdgcn_wmma_f32_16x16x32_bf16(
        /*neg_a=*/false, a, /*neg_b=*/false, b,
        /*c_mod=*/(short)0, c, /*reuse_a=*/false, /*reuse_b=*/false);
}

static __device__ __forceinline__ f32x8 zero8() {
    f32x8 z = {0.f,0.f,0.f,0.f,0.f,0.f,0.f,0.f};
    return z;
}

// ---------------------------------------------------------------------------
// fp32 -> bf16 elementwise convert
// ---------------------------------------------------------------------------
__global__ void cvt_f32_bf16(const float* __restrict__ in, bf16* __restrict__ out, int n) {
    int i = blockIdx.x * blockDim.x + threadIdx.x;
    if (i < n) out[i] = (bf16)in[i];
}

// ---------------------------------------------------------------------------
// WMMA GEMM: C[M,N] = A[M,K](bf16, row-major) * W[N,K](bf16, row-major)^T + bias
// Block = 256 threads = 8 waves (4 x 2), block tile 128x128, wave tile 32x64.
// MODE 0: scatter into Q[L,HD], K[L,HD], Vt[HD,L] bf16 (QKV projection)
// MODE 1: plain f32 store with bias (output projection)
// A-frag (16x32, 16-bit): lane l holds row m=l&15; elems 0..7 -> k=up*8+i,
//   elems 8..15 -> k=16+up*8+(i-8)  (up = l>=16).
// B-frag (32x16, 16-bit): lane l holds col n=l&15; same k order as A-frag.
// C/D (16x16 f32): lane l holds col n=l&15; vgpr r -> row m=r+up*8.
// ---------------------------------------------------------------------------
template<int MODE>
__global__ void gemm_wmma(const bf16* __restrict__ A, const bf16* __restrict__ W,
                          const float* __restrict__ bias, float* __restrict__ outp,
                          int M, int N, int K,
                          bf16* __restrict__ qb, bf16* __restrict__ kb,
                          bf16* __restrict__ vtb) {
    const int lane = threadIdx.x & 31;
    const int wave = threadIdx.x >> 5;
    const int lh   = lane & 15;
    const int up   = (lane >> 4) & 1;
    const int mbase = blockIdx.y * 128 + (wave & 3) * 32;
    const int nbase = blockIdx.x * 128 + (wave >> 2) * 64;

    f32x8 acc[2][4];
    #pragma unroll
    for (int mt = 0; mt < 2; ++mt)
        #pragma unroll
        for (int nt = 0; nt < 4; ++nt) acc[mt][nt] = zero8();

    for (int kk = 0; kk < K; kk += 32) {
        bf16x16 af[2];
        #pragma unroll
        for (int mt = 0; mt < 2; ++mt) {
            const bf16* p = A + (size_t)(mbase + mt*16 + lh) * K + kk + up*8;
            af[mt] = cat8(*(const bf16x8*)p, *(const bf16x8*)(p + 16));
            __builtin_prefetch(p + 32, 0, 3);   // next k-tile (speculative)
        }
        bf16x16 bfr[4];
        #pragma unroll
        for (int nt = 0; nt < 4; ++nt) {
            const bf16* p = W + (size_t)(nbase + nt*16 + lh) * K + kk + up*8;
            bfr[nt] = cat8(*(const bf16x8*)p, *(const bf16x8*)(p + 16));
            __builtin_prefetch(p + 32, 0, 3);   // next k-tile (speculative)
        }
        #pragma unroll
        for (int mt = 0; mt < 2; ++mt)
            #pragma unroll
            for (int nt = 0; nt < 4; ++nt)
                acc[mt][nt] = wmma_bf16(af[mt], bfr[nt], acc[mt][nt]);
    }

    #pragma unroll
    for (int mt = 0; mt < 2; ++mt) {
        #pragma unroll
        for (int nt = 0; nt < 4; ++nt) {
            const int n = nbase + nt*16 + lh;
            const float bv = bias[n];
            #pragma unroll
            for (int r = 0; r < 8; ++r) {
                const int m = mbase + mt*16 + r + up*8;
                const float v = acc[mt][nt][r] + bv;
                if (MODE == 1) {
                    outp[(size_t)m * N + n] = v;
                } else {
                    const int b   = m >> 11;        // / 2048
                    const int lq  = m & 2047;
                    const int s   = n / DD;         // 0:q 1:k 2:v
                    const int rem = n - s * DD;
                    const int h   = rem >> 6;
                    const int hd  = rem & 63;
                    const bf16 bw = (bf16)v;
                    const size_t bh = (size_t)(b * HH + h);
                    if (s == 0)      qb [(bh * LL + lq) * HD + hd] = bw;
                    else if (s == 1) kb [(bh * LL + lq) * HD + hd] = bw;
                    else             vtb[(bh * HD + hd) * LL + lq] = bw; // transposed V
                }
            }
        }
    }
}

// ---------------------------------------------------------------------------
// Flash attention. Grid: (L/64, B*H), block = 128 threads = 4 waves.
// Each wave owns 16 query rows. Scores computed transposed: S^T = Kblk * Q^T,
// so the f32 D-tiles (key rows in VGPRs, query cols across lanes) feed the
// P*V A-fragments directly, and softmax-over-keys is per-lane + one xor-16.
// All K and V fragments of a key block are loaded up front so WMMA waits are
// partial and the V loads overlap the softmax VALU work.
// ---------------------------------------------------------------------------
__global__ void flash_attn(const bf16* __restrict__ qb, const bf16* __restrict__ kb,
                           const bf16* __restrict__ vtb, const int* __restrict__ mask,
                           bf16* __restrict__ aoutb) {
    const int lane = threadIdx.x & 31;
    const int wave = threadIdx.x >> 5;
    const int lh   = lane & 15;
    const int up   = (lane >> 4) & 1;
    const int bh = blockIdx.y;
    const int b  = bh / HH;
    const int h  = bh - b * HH;
    const int q0 = blockIdx.x * 64 + wave * 16;

    const bf16* Q  = qb  + (size_t)bh * LL * HD;
    const bf16* Kp = kb  + (size_t)bh * LL * HD;
    const bf16* Vt = vtb + (size_t)bh * HD * LL;
    const int*  mk = mask + b * LL;

    // Q^T B-fragments (hd x query), held for the whole loop.
    bf16x16 qf[2];
    #pragma unroll
    for (int s = 0; s < 2; ++s) {
        const bf16* p = Q + (size_t)(q0 + lh) * HD + s*32 + up*8;
        qf[s] = cat8(*(const bf16x8*)p, *(const bf16x8*)(p + 16));
    }

    f32x8 o[4];
    #pragma unroll
    for (int t = 0; t < 4; ++t) o[t] = zero8();
    float m_run = -1e30f, l_run = 0.f;

    for (int k0 = 0; k0 < LL; k0 += 32) {
        // ---- load ALL fragments for this 32-key block first ----
        bf16x16 kf[2][2];
        #pragma unroll
        for (int j = 0; j < 2; ++j)
            #pragma unroll
            for (int s = 0; s < 2; ++s) {
                const bf16* p = Kp + (size_t)(k0 + j*16 + lh) * HD + s*32 + up*8;
                kf[j][s] = cat8(*(const bf16x8*)p, *(const bf16x8*)(p + 16));
                __builtin_prefetch(p + 32 * HD, 0, 3);   // next key block
            }
        bf16x16 vf[4];
        #pragma unroll
        for (int t = 0; t < 4; ++t) {
            const bf16* p = Vt + (size_t)(t*16 + lh) * LL + k0 + up*8;
            vf[t] = cat8(*(const bf16x8*)p, *(const bf16x8*)(p + 16));
            __builtin_prefetch(p + 32, 0, 3);            // next key block
        }

        // ---- S^T = Kblk * Q^T ----
        f32x8 st[2];
        #pragma unroll
        for (int j = 0; j < 2; ++j) {
            f32x8 a = zero8();
            a = wmma_bf16(kf[j][0], qf[0], a);
            a = wmma_bf16(kf[j][1], qf[1], a);
            st[j] = a;
        }

        // scale + mask + tile max (per query column)
        float tmax = -1e30f;
        #pragma unroll
        for (int j = 0; j < 2; ++j)
            #pragma unroll
            for (int r = 0; r < 8; ++r) {
                const int key = k0 + j*16 + r + up*8;
                float v = st[j][r] * SCL;
                if (mk[key] == 0) v = -1e30f;
                st[j][r] = v;
                tmax = fmaxf(tmax, v);
            }
        tmax = fmaxf(tmax, __shfl_xor(tmax, 16, 32));

        const float m_new = fmaxf(m_run, tmax);
        const float corr  = __expf(m_run - m_new);
        float psum = 0.f;
        #pragma unroll
        for (int j = 0; j < 2; ++j)
            #pragma unroll
            for (int r = 0; r < 8; ++r) {
                const float e = __expf(st[j][r] - m_new);
                st[j][r] = e;
                psum += e;
            }
        psum += __shfl_xor(psum, 16, 32);
        l_run = l_run * corr + psum;
        m_run = m_new;

        // rescale O accumulators: row m = r + up*8; corr[m] lives in lane m
        #pragma unroll
        for (int r = 0; r < 8; ++r) {
            const float cr = __shfl(corr, r + up*8, 32);
            #pragma unroll
            for (int t = 0; t < 4; ++t) o[t][r] *= cr;
        }

        // Assemble P A-fragment (queries x 32 keys) straight from S^T tiles.
        bf16x16 pa;
        #pragma unroll
        for (int i = 0; i < 8; ++i) {
            pa[i]     = (bf16)st[0][i];
            pa[i + 8] = (bf16)st[1][i];
        }

        // O += P * V  (V fragments already in registers)
        #pragma unroll
        for (int t = 0; t < 4; ++t)
            o[t] = wmma_bf16(pa, vf[t], o[t]);
    }

    // normalize and store [B, L, H*HD] bf16
    #pragma unroll
    for (int r = 0; r < 8; ++r) {
        const float lr  = __shfl(l_run, r + up*8, 32);
        const float inv = 1.0f / lr;
        const int qrow  = q0 + r + up*8;
        #pragma unroll
        for (int t = 0; t < 4; ++t)
            aoutb[(size_t)(b * LL + qrow) * DD + h * HD + t*16 + lh] = (bf16)(o[t][r] * inv);
    }
}

// ---------------------------------------------------------------------------
// Launch
// ---------------------------------------------------------------------------
extern "C" void kernel_launch(void* const* d_in, const int* in_sizes, int n_in,
                              void* d_out, int out_size, void* d_ws, size_t ws_size,
                              hipStream_t stream) {
    const float* x    = (const float*)d_in[0];
    const int*   mask = (const int*)  d_in[1];
    const float* Wqkv = (const float*)d_in[2];
    const float* bqkv = (const float*)d_in[3];
    const float* Wout = (const float*)d_in[4];
    const float* bout = (const float*)d_in[5];
    float* out = (float*)d_out;

    char* ws = (char*)d_ws;
    const size_t NX  = (size_t)BB * LL * DD;       // 3,145,728
    const size_t NW1 = (size_t)3 * DD * DD;        // 1,769,472
    const size_t NW2 = (size_t)DD * DD;            //   589,824
    size_t off = 0;
    bf16* xb    = (bf16*)(ws + off); off += NX  * 2;
    bf16* wqkvb = (bf16*)(ws + off); off += NW1 * 2;
    bf16* woutb = (bf16*)(ws + off); off += NW2 * 2;
    bf16* qb    = (bf16*)(ws + off); off += NX  * 2;
    bf16* kb    = (bf16*)(ws + off); off += NX  * 2;
    bf16* vtb   = (bf16*)(ws + off); off += NX  * 2;
    bf16* aoutb = (bf16*)(ws + off); off += NX  * 2;   // ~36 MB total

    // 1) convert inputs/weights to bf16
    cvt_f32_bf16<<<(int)((NX  + 255) / 256), 256, 0, stream>>>(x,    xb,    (int)NX);
    cvt_f32_bf16<<<(int)((NW1 + 255) / 256), 256, 0, stream>>>(Wqkv, wqkvb, (int)NW1);
    cvt_f32_bf16<<<(int)((NW2 + 255) / 256), 256, 0, stream>>>(Wout, woutb, (int)NW2);

    // 2) QKV projection: [4096,768] x [768,2304] -> Q/K/Vt bf16 layouts
    gemm_wmma<0><<<dim3(3 * DD / 128, BB * LL / 128), 256, 0, stream>>>(
        xb, wqkvb, bqkv, nullptr, BB * LL, 3 * DD, DD, qb, kb, vtb);

    // 3) flash attention -> aoutb [B,L,D] bf16
    flash_attn<<<dim3(LL / 64, BB * HH), 128, 0, stream>>>(qb, kb, vtb, mask, aoutb);

    // 4) output projection: [4096,768] x [768,768] -> d_out f32
    gemm_wmma<1><<<dim3(DD / 128, BB * LL / 128), 256, 0, stream>>>(
        aoutb, woutb, bout, out, BB * LL, DD, DD, nullptr, nullptr, nullptr);
}
